// SoftAttention_64381559767591
// MI455X (gfx1250) — matrix operand order repaired and checked by reading, not compiled, not run on hardware
//
#include <hip/hip_runtime.h>
#include <math.h>

typedef __attribute__((ext_vector_type(16))) _Float16 v16h;
typedef __attribute__((ext_vector_type(4)))  _Float16 h4;
typedef __attribute__((ext_vector_type(8)))  float    v8f;

#define TQ 4096
#define TVK 4096
#define DD 64
#define BM 64          // query rows per workgroup (16 per wave)
#define BN 64          // key block
#define NWAVES 4
#define KPAD 72        // LDS row stride (halves); 144B rows, 8B aligned
#define PPAD 72

__global__ __launch_bounds__(128)
void fa_causal_f16wmma(const float* __restrict__ Q,
                       const float* __restrict__ V,
                       const float* __restrict__ K,
                       const unsigned char* __restrict__ qmask,
                       const unsigned char* __restrict__ vmask,
                       const float* __restrict__ scale_p,
                       float* __restrict__ Out)
{
    // Ping-pong K/V tiles: [buf][ K-tile | V-tile ]
    __shared__ _Float16 sKV[2][2 * BN * KPAD];
    __shared__ _Float16 sP[NWAVES][16 * PPAD];

    const int tid  = threadIdx.x;
    const int lane = tid & 31;
    const int wave = tid >> 5;
    const int l16  = lane & 15;
    const int hi   = lane >> 4;            // which 16-lane half of the wave
    const int b    = blockIdx.y;
    const int blockQ = blockIdx.x * BM;
    const float scale = scale_p[0];

    const float* Kb = &K[(size_t)b * TVK * DD];
    const float* Vb = &V[(size_t)b * TVK * DD];

    // ---- Cooperative stage: f32 K/V block -> f16 LDS tiles (float4 loads) ----
    auto stage = [&](int kb, int buf) {
        const int keyBase = kb * BN;
        const float4* Kg = (const float4*)&Kb[(size_t)keyBase * DD];
        const float4* Vg = (const float4*)&Vb[(size_t)keyBase * DD];
        _Float16* dK = &sKV[buf][0];
        _Float16* dV = &sKV[buf][BN * KPAD];
#pragma unroll
        for (int i = tid; i < BN * DD / 4; i += 128) {
            int row = i >> 4, c4 = (i & 15) * 4;
            float4 kv = Kg[i];
            float4 vv = Vg[i];
            h4 hk = { (_Float16)kv.x, (_Float16)kv.y, (_Float16)kv.z, (_Float16)kv.w };
            h4 hv = { (_Float16)vv.x, (_Float16)vv.y, (_Float16)vv.z, (_Float16)vv.w };
            *(h4*)&dK[row * KPAD + c4] = hk;
            *(h4*)&dV[row * KPAD + c4] = hv;
        }
    };

    // ---- Load this wave's 16x64 Q tile once, into WMMA A-layout (f16) ----
    // A operand, contiguous k order used here (16x32 f16): lane holds row m = lane%16; half h maps to
    // k = h + 16*hi within a 32-wide K chunk.
    const int qRowA = blockQ + wave * 16 + l16;
    v16h qa[2];
#pragma unroll
    for (int ch = 0; ch < 2; ++ch) {
#pragma unroll
        for (int h = 0; h < 16; ++h) {
            int d = h + 16 * hi + 32 * ch;
            float qv = Q[((size_t)b * TQ + qRowA) * DD + d] * scale;
            qa[ch][h] = (_Float16)qv;
        }
    }

    v8f acc[4];
    float mrun[8], lrun[8];
#pragma unroll
    for (int t = 0; t < 4; ++t)
#pragma unroll
        for (int r = 0; r < 8; ++r) acc[t][r] = 0.0f;
#pragma unroll
    for (int r = 0; r < 8; ++r) { mrun[r] = -3.0e38f; lrun[r] = 0.0f; }

    const int nkb = blockIdx.x + 1;   // causal: keys only up to blockQ+63

    stage(0, 0);                      // prologue fill of buffer 0
    __syncthreads();

    for (int kb = 0; kb < nkb; ++kb) {
        const int cur = kb & 1;
        const int keyBase = kb * BN;

        // Stage next block into the other buffer while we compute on `cur`.
        if (kb + 1 < nkb) {
            stage(kb + 1, cur ^ 1);
            if (kb + 2 < nkb) {       // gfx1250 global_prefetch_b8, 2 blocks ahead
                __builtin_prefetch(&Kb[(size_t)(keyBase + 2 * BN) * DD] + tid * 32, 0, 1);
                __builtin_prefetch(&Vb[(size_t)(keyBase + 2 * BN) * DD] + tid * 32, 0, 1);
            }
        }

        const _Float16* cK = &sKV[cur][0];
        const _Float16* cV = &sKV[cur][BN * KPAD];

        // ---- Preload ALL S-phase B operands (16x ds_load_b128 issued up
        // front; in-order DS returns let the scheduler use partial dscnt
        // waits instead of gating every WMMA on dscnt==0). ----
        // B operand, contiguous k order (32x16 f16): lane holds col n = lane%16; half h -> k = h + 16*hi.
        v16h bmS[4][2];
#pragma unroll
        for (int t = 0; t < 4; ++t)
#pragma unroll
            for (int ch = 0; ch < 2; ++ch)
#pragma unroll
                for (int h = 0; h < 16; ++h) {
                    int d = h + 16 * hi + 32 * ch;
                    bmS[t][ch][h] = cK[(t * 16 + l16) * KPAD + d];
                }

        // ---- S = Q * K^T : 4 tiles of 16 keys, K-dim = D = 64 (2 WMMA each) ----
        v8f S[4];
#pragma unroll
        for (int t = 0; t < 4; ++t) {
            v8f c = {};
            c = __builtin_amdgcn_wmma_f32_16x16x32_f16(
                    false, qa[0], false, bmS[t][0], (short)0, c, false, false);
            c = __builtin_amdgcn_wmma_f32_16x16x32_f16(
                    false, qa[1], false, bmS[t][1], (short)0, c, false, false);
            S[t] = c;
        }

        // ---- Additive -1e9 masking: v_mask AND causal (matches reference) ----
#pragma unroll
        for (int t = 0; t < 4; ++t) {
            int key = keyBase + t * 16 + l16;
            float vm_pen = vmask[(size_t)b * TVK + key] ? 0.0f : -1.0e9f;
#pragma unroll
            for (int r = 0; r < 8; ++r) {
                int qRow = blockQ + wave * 16 + r + 8 * hi;   // C-layout row
                S[t][r] += (key <= qRow) ? vm_pen : -1.0e9f;
            }
        }

        // ---- Online softmax (row reductions across the 16-lane half-wave) ----
#pragma unroll
        for (int r = 0; r < 8; ++r) {
            float m = fmaxf(fmaxf(S[0][r], S[1][r]), fmaxf(S[2][r], S[3][r]));
#pragma unroll
            for (int off = 8; off >= 1; off >>= 1)
                m = fmaxf(m, __shfl_xor(m, off, 32));
            float mnew = fmaxf(mrun[r], m);
            float cs = __expf(mrun[r] - mnew);
            float rs = 0.0f;
#pragma unroll
            for (int t = 0; t < 4; ++t) {
                float p = __expf(S[t][r] - mnew);
                S[t][r] = p;
                rs += p;
            }
#pragma unroll
            for (int off = 1; off <= 8; off <<= 1)
                rs += __shfl_xor(rs, off, 32);
            lrun[r] = lrun[r] * cs + rs;
            mrun[r] = mnew;
#pragma unroll
            for (int t = 0; t < 4; ++t) acc[t][r] *= cs;
        }

        // ---- P (C-layout regs) -> per-wave LDS tile ----
        _Float16* myP = &sP[wave][0];
#pragma unroll
        for (int t = 0; t < 4; ++t)
#pragma unroll
            for (int r = 0; r < 8; ++r)
                myP[(r + 8 * hi) * PPAD + t * 16 + l16] = (_Float16)S[t][r];
        // sP is per-wave: LDS ops are in-order within a wave; wave-local wait
        // + compiler barrier is enough (no workgroup barrier needed here).
        asm volatile("s_wait_dscnt 0x0" ::: "memory");

        // ---- Preload all PV-phase operands (P in A-layout, V in B-layout),
        // then run the 8-WMMA chain. ----
        v16h pa[2];
        v16h vb[4][2];
#pragma unroll
        for (int ch = 0; ch < 2; ++ch)
#pragma unroll
            for (int h = 0; h < 16; ++h) {
                int ka = h + 16 * hi + 32 * ch;  // A-layout key
                pa[ch][h] = myP[l16 * PPAD + ka];
            }
#pragma unroll
        for (int t = 0; t < 4; ++t)
#pragma unroll
            for (int ch = 0; ch < 2; ++ch)
#pragma unroll
                for (int h = 0; h < 16; ++h) {
                    int kbk = h + 16 * hi + 32 * ch;                // B-layout key
                    vb[t][ch][h] = cV[kbk * KPAD + t * 16 + l16];
                }

        // ---- O += P * V : 4 output tiles over D, key-dim 64 (2 WMMA each) ----
#pragma unroll
        for (int t = 0; t < 4; ++t) {
            v8f c = acc[t];
            c = __builtin_amdgcn_wmma_f32_16x16x32_f16(
                    false, pa[0], false, vb[t][0], (short)0, c, false, false);
            c = __builtin_amdgcn_wmma_f32_16x16x32_f16(
                    false, pa[1], false, vb[t][1], (short)0, c, false, false);
            acc[t] = c;
        }

        // Single barrier per iteration: after it, `cur` may be overwritten
        // (at kb+2's staging) and `cur^1` staging is complete for kb+1.
        __syncthreads();
    }

    // ---- Normalize, apply q_mask, write out (C-layout rows) ----
#pragma unroll
    for (int r = 0; r < 8; ++r) {
        int qRow = blockQ + wave * 16 + r + 8 * hi;
        float qm  = qmask[(size_t)b * TQ + qRow] ? 1.0f : 0.0f;
        float inv = qm / lrun[r];
#pragma unroll
        for (int t = 0; t < 4; ++t)
            Out[((size_t)b * TQ + qRow) * DD + t * 16 + l16] = acc[t][r] * inv;
    }
}

extern "C" void kernel_launch(void* const* d_in, const int* in_sizes, int n_in,
                              void* d_out, int out_size, void* d_ws, size_t ws_size,
                              hipStream_t stream) {
    // setup_inputs order: query, value, key, q_mask, v_mask, scale
    const float* Q = (const float*)d_in[0];
    const float* V = (const float*)d_in[1];
    const float* K = (const float*)d_in[2];
    const unsigned char* qm = (const unsigned char*)d_in[3];
    const unsigned char* vm = (const unsigned char*)d_in[4];
    const float* sc = (const float*)d_in[5];
    float* out = (float*)d_out;

    const int B = in_sizes[0] / (TQ * DD);
    dim3 grid(TQ / BM, B);
    fa_causal_f16wmma<<<grid, dim3(128), 0, stream>>>(Q, V, K, qm, vm, sc, out);
}
